// TemporalGraphTransformer_24043226923415
// MI455X (gfx1250) — matrix operand order repaired and checked by reading, not compiled, not run on hardware
//
#include <hip/hip_runtime.h>
#include <hip/hip_bf16.h>

// ---------------------------------------------------------------------------
// Temporal Graph Transformer for MI455X (gfx1250), bf16 WMMA everywhere.
// B=1, N=512, D=128, H=4, DK=32, L=2, OSC=4.
//
// d_in layout assumption (JAX pytree flatten, dict keys sorted, ASCII order):
//   0 node_features  1 adjacency  2 edge_times  3 node_phases
//   then per layer (stride 30):
//   attn: +0 Wk +1 Wo +2 Wq +3 Wt +4 Wv +5 bk +6 bo +7 bq +8 bt +9 bv
//   ffn : +10 W1 +11 W2 +12 b +13 b1 +14 b2 +15 g
//   ln1 : +16 g +17 b       ln2: +18 g +19 b
//   msg : +20 U1 +21 U2 +22 W1 +23 W2 +24 Wg +25 b1 +26 b2 +27 bg +28 ub1 +29 ub2
// ---------------------------------------------------------------------------

typedef __attribute__((ext_vector_type(16))) __bf16          v16bf;
typedef __attribute__((ext_vector_type(16))) unsigned short  v16us;
typedef __attribute__((ext_vector_type(8)))  unsigned short  v8us;
typedef __attribute__((ext_vector_type(8)))  float           v8f;

#define NNODE 512
#define DIMF  128

__device__ __forceinline__ unsigned short f2bf(float f) {
  unsigned u = __float_as_uint(f);
  u += 0x7FFFu + ((u >> 16) & 1u);   // round-to-nearest-even
  return (unsigned short)(u >> 16);
}

__device__ __forceinline__ float fast_rcp(float x) {
  return __builtin_amdgcn_rcpf(x);   // v_rcp_f32, no IEEE div expansion
}

// Within a 32-wide K group, permute columns so each lane-half's WMMA A
// fragment (logical K = {kh*8..+7, 16+kh*8..+7}) is one contiguous 32B run
// at column kh*16.  perm: subgroup 0->0, 1->2, 2->1, 3->3.
__device__ __forceinline__ int a_perm(int c) {
  int g  = (c >> 3) & 3;
  int gp = ((g & 1) << 1) | (g >> 1);
  return (c & ~31) | (gp << 3) | (c & 7);
}

__device__ __forceinline__ v8f wmma_bf16(const v16us& a, const v16us& b, v8f c) {
  return __builtin_amdgcn_wmma_f32_16x16x32_bf16(
      false, __builtin_bit_cast(v16bf, a),
      false, __builtin_bit_cast(v16bf, b),
      (short)0, c, false, false);
}

// async copy of 16 bytes global -> LDS (gfx1250, ASYNCcnt-tracked)
__device__ __forceinline__ void async_g2l_b128(unsigned lds_off, const void* gaddr) {
  unsigned long long ga = (unsigned long long)(uintptr_t)gaddr;
  asm volatile("global_load_async_to_lds_b128 %0, %1, off"
               :: "v"(lds_off), "v"(ga) : "memory");
}
__device__ __forceinline__ void async_wait0() {
  asm volatile("s_wait_asynccnt 0" ::: "memory");
}
__device__ __forceinline__ unsigned lds_addr32(const void* p) {
  return (unsigned)(uintptr_t)p;   // low 32 bits of flat LDS aperture == LDS offset
}

// ---------------------------------------------------------------------------
// Tiled GEMM:  C[M,N] = act(scale * (A[M,K] @ op(W)^T) + bias) (+resid)
//   transB==0: W is (N,K) row-major, row stride ldw  (torch-style weight)
//   transB==1: W is (K,N) row-major, row stride ldw
// Block: 128 threads = 4 waves; block tile 32(M) x 64(N); per-wave two 16x16
// tiles sharing one B fragment. act: 0=none 1=relu 2=gelu(exact)
// ---------------------------------------------------------------------------
template <int K>
__global__ __launch_bounds__(128)
void gemm_bf16_kernel(const float* __restrict__ A, int lda,
                      const float* __restrict__ W, int ldw, int transB,
                      const float* __restrict__ bias,
                      const float* __restrict__ resid,
                      float* __restrict__ C, int ldc,
                      int M, int N, float scale, int act)
{
  __shared__ alignas(32) unsigned short As[32][32];       // 2 KB, A-permuted
  __shared__ alignas(32) unsigned short Bs[4][16][32];    // 4 KB, n-major per tile

  const int tid  = threadIdx.x;
  const int wave = tid >> 5;
  const int lane = tid & 31;
  const int m0   = blockIdx.x * 32;
  const int nblk = blockIdx.y * 64;
  const int n0   = nblk + wave * 16;
  const bool active = (n0 < N);
  const int mm = lane & 15;
  const int kh = lane >> 4;

  v8f acc0, acc1;
#pragma unroll
  for (int z = 0; z < 8; ++z) { acc0[z] = 0.0f; acc1[z] = 0.0f; }

  for (int k0 = 0; k0 < K; k0 += 32) {
    // stage A tile (32 x 32), fp32 -> bf16, WMMA-A column permutation
    for (int e = tid; e < 1024; e += 128) {
      int rr = e >> 5, c = e & 31;
      As[rr][a_perm(c)] = f2bf(A[(size_t)(m0 + rr) * lda + k0 + c]);
    }
    // stage B tiles, n-major so fragments are contiguous
    for (int e = tid; e < 2048; e += 128) {
      int rr = e & 31;          // k local
      int cc = (e >> 5) & 15;   // n local
      int t  = e >> 9;          // wave tile
      int n  = nblk + t * 16 + cc;
      float w = 0.0f;
      if (n < N)
        w = transB ? W[(size_t)(k0 + rr) * ldw + n]
                   : W[(size_t)n * ldw + k0 + rr];
      Bs[t][cc][a_perm(rr)] = f2bf(w);
    }
    if (k0 + 32 < K)
      __builtin_prefetch(&A[(size_t)m0 * lda + k0 + 32]);
    __syncthreads();

    if (active) {
      // A fragments: one contiguous 32B run per lane (permuted layout)
      v16us av0 = *(const v16us*)&As[mm][kh * 16];
      v16us av1 = *(const v16us*)&As[16 + mm][kh * 16];
      // B fragment: one contiguous 32B run per lane
      v16us bv = *(const v16us*)&Bs[wave][mm][kh * 16];
      acc0 = wmma_bf16(av0, bv, acc0);
      acc1 = wmma_bf16(av1, bv, acc1);
    }
    __syncthreads();
  }

  if (active) {
    const int nn = n0 + mm;
    const float bval = bias ? bias[nn] : 0.0f;
#pragma unroll
    for (int v = 0; v < 8; ++v) {
      int mr = m0 + v + (kh << 3);
      float x = acc0[v] * scale + bval;
      if (act == 1)      x = fmaxf(x, 0.0f);
      else if (act == 2) x = 0.5f * x * (1.0f + erff(x * 0.70710678118654752f));
      if (resid) x += resid[(size_t)mr * ldc + nn];
      C[(size_t)mr * ldc + nn] = x;

      int mr2 = mr + 16;
      float y = acc1[v] * scale + bval;
      if (act == 1)      y = fmaxf(y, 0.0f);
      else if (act == 2) y = 0.5f * y * (1.0f + erff(y * 0.70710678118654752f));
      if (resid) y += resid[(size_t)mr2 * ldc + nn];
      C[(size_t)mr2 * ldc + nn] = y;
    }
  }
}

// ---------------------------------------------------------------------------
// LayerNorm over D=128, one block per row.
// ---------------------------------------------------------------------------
__global__ __launch_bounds__(128)
void layernorm_kernel(const float* __restrict__ x, const float* __restrict__ g,
                      const float* __restrict__ b, float* __restrict__ y)
{
  const int i = blockIdx.x, f = threadIdx.x;
  __shared__ float red[128];
  float v = x[i * 128 + f];
  red[f] = v; __syncthreads();
  for (int s = 64; s > 0; s >>= 1) { if (f < s) red[f] += red[f + s]; __syncthreads(); }
  float mean = red[0] * (1.0f / 128.0f); __syncthreads();
  float d = v - mean;
  red[f] = d * d; __syncthreads();
  for (int s = 64; s > 0; s >>= 1) { if (f < s) red[f] += red[f + s]; __syncthreads(); }
  float var = red[0] * (1.0f / 128.0f);
  y[i * 128 + f] = d * rsqrtf(var + 1e-5f) * g[f] + b[f];
}

// ---------------------------------------------------------------------------
// Attention softmax with adjacency mask + temporal recency bias.
// grid (N, H), block 256. scores layout: [h][i][j], j stride 1.
// ---------------------------------------------------------------------------
__global__ __launch_bounds__(256)
void attn_softmax_kernel(float* __restrict__ scores,
                         const float* __restrict__ adj,
                         const float* __restrict__ et,
                         const float* __restrict__ Wt,
                         const float* __restrict__ bt)
{
  const int i = blockIdx.x, h = blockIdx.y, t = threadIdx.x;
  float* row = scores + ((size_t)h * NNODE + i) * NNODE;
  const float wt = Wt[h], btv = bt[h];
  __shared__ float red[256];
  float l[2];
  float lmax = -1e30f;
#pragma unroll
  for (int jj = 0; jj < 2; ++jj) {
    int j = t + jj * 256;
    float a  = adj[i * NNODE + j];
    float tw = __expf(-0.1f * fmaxf(1.0f - et[i * NNODE + j], 0.0f));
    float v  = (a == 0.0f) ? -1e30f : (row[j] + tw * wt + btv);
    l[jj] = v;
    lmax = fmaxf(lmax, v);
  }
  red[t] = lmax; __syncthreads();
  for (int s = 128; s > 0; s >>= 1) { if (t < s) red[t] = fmaxf(red[t], red[t + s]); __syncthreads(); }
  lmax = red[0]; __syncthreads();
  float lsum = 0.0f;
#pragma unroll
  for (int jj = 0; jj < 2; ++jj) { l[jj] = __expf(l[jj] - lmax); lsum += l[jj]; }
  red[t] = lsum; __syncthreads();
  for (int s = 128; s > 0; s >>= 1) { if (t < s) red[t] += red[t + s]; __syncthreads(); }
  float inv = fast_rcp(red[0]);
#pragma unroll
  for (int jj = 0; jj < 2; ++jj) row[t + jj * 256] = l[jj] * inv;
}

// ---------------------------------------------------------------------------
__global__ __launch_bounds__(128)
void counts_kernel(const float* __restrict__ adj, float* __restrict__ counts)
{
  const int i = blockIdx.x, t = threadIdx.x;
  float s = 0.0f;
  for (int j = t; j < NNODE; j += 128) s += (adj[i * NNODE + j] != 0.0f) ? 1.0f : 0.0f;
  __shared__ float red[128];
  red[t] = s; __syncthreads();
  for (int k = 64; k > 0; k >>= 1) { if (t < k) red[t] += red[t + k]; __syncthreads(); }
  if (t == 0) counts[i] = fmaxf(red[0], 1.0f);
}

__global__ __launch_bounds__(256)
void bf16_convert_kernel(const float* __restrict__ src, unsigned short* __restrict__ dst, int n)
{
  int e = blockIdx.x * 256 + threadIdx.x;
  if (e < n) dst[e] = f2bf(src[e]);
}

// ---------------------------------------------------------------------------
// Pair-wise gated message passing.
// Block = 16 receivers (i) x 8 senders (j); 128 threads = 4 waves.
//   h[p=il*8+jl, f] = relu(r[i,f] + s[j,f] + b1[f])      (128x128 bf16 in LDS)
//   msgs = h @ W2^T + b2 ; gate = sigmoid(Wg . cos(dphi) + bg)
//   agg[i] += sum_j mask(i,j)*gate*msgs / counts[i]       (global atomic)
// r/s tiles staged with gfx1250 async global->LDS copies. B fragments are
// hoisted per n-tile and reused across all 8 m-tiles (8 WMMAs per B load).
// hS stored in contiguous k order (as w2t) -> single v16us fragment loads.
// ---------------------------------------------------------------------------
__global__ __launch_bounds__(128)
void pair_msg_kernel(const float* __restrict__ r, const float* __restrict__ s,
                     const unsigned short* __restrict__ w2t,   // bf16 W2 (n,k)
                     const float* __restrict__ b1, const float* __restrict__ b2,
                     const float* __restrict__ Wg, const float* __restrict__ bg,
                     const float* __restrict__ phase, const float* __restrict__ adj,
                     const float* __restrict__ counts, float* __restrict__ agg)
{
  __shared__ alignas(32) unsigned short hS[128][128];   // 32 KB, contiguous k order
  __shared__ alignas(16) float rS[16][128];             //  8 KB
  __shared__ alignas(16) float sS[8][128];              //  4 KB
  __shared__ float cosS[16][8][4];                      //  2 KB
  __shared__ float gwS[128][4];                         //  2 KB
  __shared__ float aggS[16][128];                       //  8 KB
  __shared__ float mS[16][8];
  __shared__ float b1S[128], b2S[128], bgS[128], rcntS[16];

  const int i0 = blockIdx.x * 16;
  const int j0 = blockIdx.y * 8;
  const int tid = threadIdx.x;

  // r tile: 16*128 f32 = 8 KB contiguous; s tile: 4 KB contiguous -> async DMA
  {
    const unsigned rbase = lds_addr32(&rS[0][0]);
    const float* gr = r + (size_t)i0 * 128;
    for (int c = tid; c < 512; c += 128)
      async_g2l_b128(rbase + c * 16, gr + c * 4);
    const unsigned sbase = lds_addr32(&sS[0][0]);
    const float* gs = s + (size_t)j0 * 128;
    for (int c = tid; c < 256; c += 128)
      async_g2l_b128(sbase + c * 16, gs + c * 4);
  }
  if (tid < 128) {
    b1S[tid] = b1[tid]; b2S[tid] = b2[tid]; bgS[tid] = bg[tid];
#pragma unroll
    for (int o = 0; o < 4; ++o) gwS[tid][o] = Wg[tid * 4 + o];
  }
  if (tid < 16) rcntS[tid] = fast_rcp(counts[i0 + tid]);
  for (int e = tid; e < 16 * 8 * 4; e += 128) {
    int il = e >> 5, jl = (e >> 2) & 7, o = e & 3;
    cosS[il][jl][o] = __cosf(phase[(i0 + il) * 4 + o] - phase[(j0 + jl) * 4 + o]);
  }
  for (int e = tid; e < 16 * 8; e += 128) {
    int il = e >> 3, jl = e & 7;
    mS[il][jl] = (adj[(size_t)(i0 + il) * NNODE + (j0 + jl)] != 0.0f) ? 1.0f : 0.0f;
  }
  async_wait0();
  __syncthreads();

  // build h tile: 8 contiguous bf16 per thread-iteration, stored at the
  // contiguous k position, b128 LDS stores
  for (int c = tid; c < 2048; c += 128) {
    int p  = c >> 4;
    int f0 = (c & 15) * 8;
    int il = p >> 3, jl = p & 7;
    v8us o;
#pragma unroll
    for (int u = 0; u < 8; ++u)
      o[u] = f2bf(fmaxf(rS[il][f0 + u] + sS[jl][f0 + u] + b1S[f0 + u], 0.0f));
    *(v8us*)&hS[p][f0] = o;   // contiguous k order, matches w2t B fragment
  }
  __syncthreads();

  const int wave = tid >> 5, lane = tid & 31;
  const int mm = lane & 15, kh = lane >> 4;

  // n-tile outer: B fragments (4 K-steps) loaded once, reused by 8 m-tiles
  for (int nt = wave; nt < 8; nt += 4) {
    const int n0 = nt * 16;
    const int f  = n0 + mm;
    v16us bvs[4];
#pragma unroll
    for (int ks = 0; ks < 4; ++ks)
      bvs[ks] = *(const v16us*)(w2t + (size_t)f * 128 + ks * 32 + kh * 16);
    const float gw0 = gwS[f][0], gw1 = gwS[f][1], gw2 = gwS[f][2], gw3 = gwS[f][3];
    const float bgv = bgS[f], b2v = b2S[f];

    for (int mt = 0; mt < 8; ++mt) {
      const int m0 = mt * 16;
      v8f acc;
#pragma unroll
      for (int z = 0; z < 8; ++z) acc[z] = 0.0f;

#pragma unroll
      for (int ks = 0; ks < 4; ++ks) {
        v16us av = *(const v16us*)&hS[m0 + mm][ks * 32 + kh * 16];
        acc = wmma_bf16(av, bvs[ks], acc);
      }

      // per-lane: all 8 accumulator rows share one receiver il = 2*mt + kh
      const int il = (m0 >> 3) + kh;
      float sum = 0.0f;
#pragma unroll
      for (int v = 0; v < 8; ++v) {
        int jl = v;                           // (m0 + kh*8 + v) & 7 == v
        float val = acc[v] + b2v;
        float gl = gw0 * cosS[il][jl][0] + gw1 * cosS[il][jl][1]
                 + gw2 * cosS[il][jl][2] + gw3 * cosS[il][jl][3] + bgv;
        gl = fast_rcp(1.0f + __expf(-gl));    // sigmoid, v_rcp not IEEE div
        sum += val * gl * mS[il][jl];
      }
      aggS[il][f] = sum;   // unique (il,f) per (tile,lane) -> plain store
    }
  }
  __syncthreads();

  for (int e = tid; e < 16 * 128; e += 128) {
    int il = e >> 7, f = e & 127;
    unsafeAtomicAdd(&agg[(size_t)(i0 + il) * 128 + f], aggS[il][f] * rcntS[il]);
  }
}

// ---------------------------------------------------------------------------
__global__ __launch_bounds__(256)
void concat_kernel(const float* __restrict__ xn, const float* __restrict__ agg,
                   float* __restrict__ cat)
{
  int i = blockIdx.x, c = threadIdx.x;
  cat[(size_t)i * 256 + c] = (c < 128) ? xn[(size_t)i * 128 + c]
                                       : agg[(size_t)i * 128 + (c - 128)];
}

__global__ __launch_bounds__(256)
void copy_out_kernel(const float* __restrict__ x, float* __restrict__ out)
{
  int e = blockIdx.x * 256 + threadIdx.x;
  out[e] = x[e];
}

__global__ __launch_bounds__(128)
void mean_kernel(const float* __restrict__ x, float* __restrict__ out)
{
  const int f = blockIdx.x, t = threadIdx.x;
  float s = 0.0f;
  for (int i = t; i < NNODE; i += 128) s += x[(size_t)i * 128 + f];
  __shared__ float red[128];
  red[t] = s; __syncthreads();
  for (int k = 64; k > 0; k >>= 1) { if (t < k) red[t] += red[t + k]; __syncthreads(); }
  if (t == 0) out[f] = red[0] * (1.0f / 512.0f);
}

// ---------------------------------------------------------------------------
static inline void launch_gemm(hipStream_t st,
                               const float* A, int lda,
                               const float* W, int ldw, int transB,
                               const float* bias, const float* resid,
                               float* C, int ldc,
                               int M, int N, int K, float scale, int act)
{
  dim3 grid(M / 32, (N + 63) / 64);
  switch (K) {
    case 32:
      gemm_bf16_kernel<32><<<grid, 128, 0, st>>>(A, lda, W, ldw, transB, bias, resid, C, ldc, M, N, scale, act);
      break;
    case 128:
      gemm_bf16_kernel<128><<<grid, 128, 0, st>>>(A, lda, W, ldw, transB, bias, resid, C, ldc, M, N, scale, act);
      break;
    case 256:
      gemm_bf16_kernel<256><<<grid, 128, 0, st>>>(A, lda, W, ldw, transB, bias, resid, C, ldc, M, N, scale, act);
      break;
    default:
      gemm_bf16_kernel<512><<<grid, 128, 0, st>>>(A, lda, W, ldw, transB, bias, resid, C, ldc, M, N, scale, act);
      break;
  }
}

extern "C" void kernel_launch(void* const* d_in, const int* in_sizes, int n_in,
                              void* d_out, int out_size, void* d_ws, size_t ws_size,
                              hipStream_t stream)
{
  (void)in_sizes; (void)n_in; (void)out_size; (void)ws_size;
  const float* nodef = (const float*)d_in[0];
  const float* adj   = (const float*)d_in[1];
  const float* etime = (const float*)d_in[2];
  const float* phase = (const float*)d_in[3];
  auto P = [&](int l, int idx) -> const float* {
    return (const float*)d_in[4 + l * 30 + idx];
  };

  float* ws     = (float*)d_ws;
  float* xbuf   = ws + 0;         // 65536
  float* xn     = ws + 65536;     // 65536
  float* q      = ws + 131072;    // 65536
  float* kbuf   = ws + 196608;    // 65536
  float* vbuf   = ws + 262144;    // 65536
  float* ao     = ws + 327680;    // 65536
  float* rbuf   = ws + 393216;    // 65536
  float* sbuf   = ws + 458752;    // 65536
  float* agg    = ws + 524288;    // 65536
  float* cat    = ws + 589824;    // 131072
  float* t1     = ws + 720896;    // 65536
  float* ffnh   = ws + 786432;    // 262144
  float* scores = ws + 1048576;   // 4*512*512 = 1048576
  float* counts = ws + 2097152;   // 512
  unsigned short* w2bf = (unsigned short*)(ws + 2097664); // 16384 bf16

  hipMemcpyAsync(xbuf, nodef, (size_t)65536 * sizeof(float),
                 hipMemcpyDeviceToDevice, stream);
  counts_kernel<<<NNODE, 128, 0, stream>>>(adj, counts);

  const float INV_SQRT_DK = 0.17677669529663687f;   // 1/sqrt(32)

  for (int l = 0; l < 2; ++l) {
    const float *Wk = P(l,0), *Wo = P(l,1), *Wq = P(l,2), *Wt = P(l,3), *Wv = P(l,4);
    const float *bk = P(l,5), *bo = P(l,6), *bq = P(l,7), *bt = P(l,8), *bv = P(l,9);
    const float *F1 = P(l,10), *F2 = P(l,11), *fbn = P(l,12), *fb1 = P(l,13),
                *fb2 = P(l,14), *fgn = P(l,15);
    const float *g1 = P(l,16), *bb1 = P(l,17), *g2 = P(l,18), *bb2 = P(l,19);
    const float *U1 = P(l,20), *U2 = P(l,21), *mW1 = P(l,22), *mW2 = P(l,23),
                *mWg = P(l,24), *mb1 = P(l,25), *mb2 = P(l,26), *mbg = P(l,27),
                *ub1 = P(l,28), *ub2 = P(l,29);

    // ---- pre-norm + dynamic graph attention ----
    layernorm_kernel<<<NNODE, 128, 0, stream>>>(xbuf, g1, bb1, xn);
    launch_gemm(stream, xn, 128, Wq, 128, 0, bq, nullptr, q,    128, 512, 128, 128, 1.0f, 0);
    launch_gemm(stream, xn, 128, Wk, 128, 0, bk, nullptr, kbuf, 128, 512, 128, 128, 1.0f, 0);
    launch_gemm(stream, xn, 128, Wv, 128, 0, bv, nullptr, vbuf, 128, 512, 128, 128, 1.0f, 0);
    for (int h = 0; h < 4; ++h)
      launch_gemm(stream, q + h * 32, 128, kbuf + h * 32, 128, 0, nullptr, nullptr,
                  scores + (size_t)h * 262144, 512, 512, 512, 32, INV_SQRT_DK, 0);
    attn_softmax_kernel<<<dim3(NNODE, 4), 256, 0, stream>>>(scores, adj, etime, Wt, bt);
    for (int h = 0; h < 4; ++h)
      launch_gemm(stream, scores + (size_t)h * 262144, 512, vbuf + h * 32, 128, 1,
                  nullptr, nullptr, ao + h * 32, 128, 512, 32, 512, 1.0f, 0);
    launch_gemm(stream, ao, 128, Wo, 128, 0, bo, xn, xbuf, 128, 512, 128, 128, 1.0f, 0);

    // ---- norm + phase-gated message passing ----
    layernorm_kernel<<<NNODE, 128, 0, stream>>>(xbuf, g2, bb2, xn);
    launch_gemm(stream, xn, 128, mW1,       256, 0, nullptr, nullptr, rbuf, 128, 512, 128, 128, 1.0f, 0);
    launch_gemm(stream, xn, 128, mW1 + 128, 256, 0, nullptr, nullptr, sbuf, 128, 512, 128, 128, 1.0f, 0);
    bf16_convert_kernel<<<64, 256, 0, stream>>>(mW2, w2bf, 16384);
    hipMemsetAsync(agg, 0, (size_t)65536 * sizeof(float), stream);
    pair_msg_kernel<<<dim3(32, 64), 128, 0, stream>>>(rbuf, sbuf, w2bf, mb1, mb2,
                                                      mWg, mbg, phase, adj, counts, agg);
    concat_kernel<<<NNODE, 256, 0, stream>>>(xn, agg, cat);
    launch_gemm(stream, cat, 256, U1, 256, 0, ub1, nullptr, t1,   128, 512, 128, 256, 1.0f, 1);
    launch_gemm(stream, t1,  128, U2, 128, 0, ub2, xn,      xbuf, 128, 512, 128, 128, 1.0f, 0);

    // ---- FFN ----
    layernorm_kernel<<<NNODE, 128, 0, stream>>>(xbuf, fgn, fbn, xn);
    launch_gemm(stream, xn,   128, F1, 128, 0, fb1, nullptr, ffnh, 512, 512, 512, 128, 1.0f, 2);
    launch_gemm(stream, ffnh, 512, F2, 512, 0, fb2, xbuf,    xbuf, 128, 512, 128, 512, 1.0f, 0);
  }

  copy_out_kernel<<<256, 256, 0, stream>>>(xbuf, (float*)d_out);
  mean_kernel<<<128, 128, 0, stream>>>(xbuf, (float*)d_out + 65536);
}
